// _MambaBlock_86251533238390
// MI455X (gfx1250) — matrix operand order repaired and checked by reading, not compiled, not run on hardware
//
#include <hip/hip_runtime.h>
#include <hip/hip_bf16.h>

// ---------------- problem constants ----------------
#define D_MODEL 1024
#define D_STATE 16
#define D_CONVK 4
#define D_INNER 2048
#define DT_RANK 64
#define B_SZ 2
#define L_SEQ 2048
#define T_TOK (B_SZ * L_SEQ)   /* 4096 tokens */
#define E2 (2 * D_INNER)       /* 4096 = x|z  */
#define XDBL 96                /* DT_RANK + 2*D_STATE */

typedef __bf16 bf16_t;
typedef bf16_t v16bf __attribute__((ext_vector_type(16)));
typedef float  v8f   __attribute__((ext_vector_type(8)));

// ---- CDNA5 async global->LDS path (guarded; fall back to plain LDS) ----
#if defined(__has_builtin)
#if __has_builtin(__builtin_amdgcn_global_load_async_to_lds_b32) && \
    __has_builtin(__builtin_amdgcn_s_wait_asynccnt)
#define MAMBA_ASYNC_LDS 1
#endif
#endif
#ifndef MAMBA_ASYNC_LDS
#define MAMBA_ASYNC_LDS 0
#endif

#if MAMBA_ASYNC_LDS
__device__ __forceinline__ void g2l_async_b32(const float* g, float* lds) {
  __builtin_amdgcn_global_load_async_to_lds_b32(
      (__attribute__((address_space(1))) int*)(uintptr_t)g,
      (__attribute__((address_space(3))) int*)lds,
      /*offset=*/0, /*cpol=*/0);
}
__device__ __forceinline__ void wait_async0() {
  __builtin_amdgcn_s_wait_asynccnt(0);
}
#endif

__device__ __forceinline__ bf16_t f2bf(float f) {
  union { float f; unsigned u; } in; in.f = f;
  unsigned u = in.u;
  u += 0x7FFFu + ((u >> 16) & 1u);           // round-to-nearest-even
  union { unsigned short s; bf16_t h; } out;
  out.s = (unsigned short)(u >> 16);
  return out.h;
}

__device__ __forceinline__ float sigmoidf_(float x) {
  return 1.0f / (1.0f + __expf(-x));
}

// ---------------- fp32 -> bf16 weight conversion ----------------
__global__ void cvt_f32_bf16_kernel(const float* __restrict__ src,
                                    bf16_t* __restrict__ dst, int n) {
  int i = blockIdx.x * blockDim.x + threadIdx.x;
  if (i < n) dst[i] = f2bf(src[i]);
}

// ---------------- LayerNorm -> bf16 activations ----------------
__global__ void ln_kernel(const float* __restrict__ seq,
                          const float* __restrict__ g,
                          const float* __restrict__ b,
                          bf16_t* __restrict__ xn) {
  __shared__ float red[256];
  __shared__ float s_mu, s_rstd;
  const int t = blockIdx.x;
  const int tid = threadIdx.x;
  const float* row = seq + (size_t)t * D_MODEL;

  float s = 0.f;
  #pragma unroll
  for (int i = 0; i < D_MODEL / 256; ++i) s += row[tid + i * 256];
  red[tid] = s; __syncthreads();
  for (int st = 128; st > 0; st >>= 1) {
    if (tid < st) red[tid] += red[tid + st];
    __syncthreads();
  }
  if (tid == 0) s_mu = red[0] * (1.0f / D_MODEL);
  __syncthreads();
  const float mu = s_mu;

  float v = 0.f;
  #pragma unroll
  for (int i = 0; i < D_MODEL / 256; ++i) {
    float d = row[tid + i * 256] - mu; v += d * d;
  }
  red[tid] = v; __syncthreads();
  for (int st = 128; st > 0; st >>= 1) {
    if (tid < st) red[tid] += red[tid + st];
    __syncthreads();
  }
  if (tid == 0) s_rstd = rsqrtf(red[0] * (1.0f / D_MODEL) + 1e-5f);
  __syncthreads();
  const float rstd = s_rstd;

  #pragma unroll
  for (int i = 0; i < D_MODEL / 256; ++i) {
    int c = tid + i * 256;
    xn[(size_t)t * D_MODEL + c] = f2bf((row[c] - mu) * rstd * g[c] + b[c]);
  }
}

// ---------------- generic bf16 WMMA GEMM, 32x32 tile per wave ----------------
// C[T,N] (fp32, row stride N) = A[T,K]_bf16 (row stride lda) x W[N,K]_bf16^T
// mode 0: plain   mode 1: softplus(C + bias[n])
// Cbf (optional): bf16 mirror of C.
// launch: block = 32 (one wave, EXEC all-ones), grid = (N/32, T/32).
// Per k-step: 2 A-frags + 2 B-frags -> 4 v_wmma (2x reuse of both operands).
__global__ void gemm_bf16_wmma_kernel(float* __restrict__ C,
                                      bf16_t* __restrict__ Cbf,
                                      const bf16_t* __restrict__ A,
                                      const bf16_t* __restrict__ W,
                                      const float* __restrict__ bias,
                                      int N, int K, int lda, int mode) {
  const int lane  = threadIdx.x & 31;
  const int lrow  = lane & 15;
  const int lhalf = lane >> 4;           // 0: lanes 0-15, 1: lanes 16-31
  const int tn = blockIdx.x;             // 32-wide N tile
  const int tm = blockIdx.y;             // 32-tall M tile

  const int n0 = tn * 32 + lrow;         // output column set 0 (this lane)
  const int n1 = n0 + 16;                // output column set 1
  const int m0 = tm * 32 + lrow;         // A row, sub-tile 0
  const int m1 = m0 + 16;                // A row, sub-tile 1

  v8f acc00 = {}, acc01 = {}, acc10 = {}, acc11 = {};

  const bf16_t* arow0 = A + (size_t)m0 * lda;
  const bf16_t* arow1 = A + (size_t)m1 * lda;
  const bf16_t* wrow0 = W + (size_t)n0 * K;
  const bf16_t* wrow1 = W + (size_t)n1 * K;

  for (int k0 = 0; k0 < K; k0 += 32) {
    // A fragment (ISA 7.12.2, 16-bit A 16x32): lane<16 -> K {0..7,16..23}
    const bf16_t* pa0 = arow0 + k0 + lhalf * 8;
    const bf16_t* pa1 = arow1 + k0 + lhalf * 8;
    v16bf a0, a1;
    #pragma unroll
    for (int i = 0; i < 8; ++i) {
      a0[i] = pa0[i];  a0[8 + i] = pa0[16 + i];
      a1[i] = pa1[i];  a1[8 + i] = pa1[16 + i];
    }
    // B fragment (32x16): same K order as A: lane<16 -> K {0..7,16..23}, lanes>=16 -> K {8..15,24..31}
    const bf16_t* pb0 = wrow0 + k0 + lhalf * 8;
    const bf16_t* pb1 = wrow1 + k0 + lhalf * 8;
    v16bf b0, b1;
    #pragma unroll
    for (int i = 0; i < 8; ++i) {
      b0[i] = pb0[i];  b0[8 + i] = pb0[16 + i];
      b1[i] = pb1[i];  b1[8 + i] = pb1[16 + i];
    }

    __builtin_prefetch(pa0 + 32, 0, 3);    // global_prefetch_b8, next K block
    __builtin_prefetch(pb0 + 32, 0, 3);
    __builtin_prefetch(pb1 + 32, 0, 3);

    acc00 = __builtin_amdgcn_wmma_f32_16x16x32_bf16(
        false, a0, false, b0, (short)0, acc00, false, false);
    acc01 = __builtin_amdgcn_wmma_f32_16x16x32_bf16(
        false, a0, false, b1, (short)0, acc01, false, false);
    acc10 = __builtin_amdgcn_wmma_f32_16x16x32_bf16(
        false, a1, false, b0, (short)0, acc10, false, false);
    acc11 = __builtin_amdgcn_wmma_f32_16x16x32_bf16(
        false, a1, false, b1, (short)0, acc11, false, false);
  }

  const float bn0 = (mode == 1 && bias) ? bias[n0] : 0.0f;
  const float bn1 = (mode == 1 && bias) ? bias[n1] : 0.0f;

  // C/D layout: lane<16 -> N=lane, VGPR r -> M=r ; lanes 16-31 -> M=r+8
  #pragma unroll
  for (int r = 0; r < 8; ++r) {
    int row0 = tm * 32 + lhalf * 8 + r;
    int row1 = row0 + 16;
    float v00 = acc00[r], v01 = acc01[r], v10 = acc10[r], v11 = acc11[r];
    if (mode == 1) {
      v00 += bn0; v01 += bn1; v10 += bn0; v11 += bn1;
      v00 = (v00 > 20.0f) ? v00 : log1pf(__expf(v00));  // softplus
      v01 = (v01 > 20.0f) ? v01 : log1pf(__expf(v01));
      v10 = (v10 > 20.0f) ? v10 : log1pf(__expf(v10));
      v11 = (v11 > 20.0f) ? v11 : log1pf(__expf(v11));
    }
    size_t i00 = (size_t)row0 * N + n0;
    size_t i01 = (size_t)row0 * N + n1;
    size_t i10 = (size_t)row1 * N + n0;
    size_t i11 = (size_t)row1 * N + n1;
    C[i00] = v00; C[i01] = v01; C[i10] = v10; C[i11] = v11;
    if (Cbf) {
      Cbf[i00] = f2bf(v00); Cbf[i01] = f2bf(v01);
      Cbf[i10] = f2bf(v10); Cbf[i11] = f2bf(v11);
    }
  }
}

// ---------------- depthwise causal conv (k=4) + SiLU ----------------
__global__ void conv_silu_kernel(const float* __restrict__ xz,
                                 const float* __restrict__ cw,
                                 const float* __restrict__ cb,
                                 float* __restrict__ xc,
                                 bf16_t* __restrict__ xcbf) {
  int idx = blockIdx.x * blockDim.x + threadIdx.x;     // t*D_INNER + d
  if (idx >= T_TOK * D_INNER) return;
  int t = idx / D_INNER;
  int d = idx - t * D_INNER;
  int b = t / L_SEQ;
  int l = t - b * L_SEQ;

  float acc = cb[d];
  #pragma unroll
  for (int k = 0; k < D_CONVK; ++k) {
    int ll = l - (D_CONVK - 1) + k;
    if (ll >= 0)
      acc += xz[((size_t)(b * L_SEQ + ll)) * E2 + d] * cw[d * D_CONVK + k];
  }
  float s = acc * sigmoidf_(acc);   // SiLU
  xc[idx] = s;
  xcbf[idx] = f2bf(s);
}

// ---------------- selective scan ----------------
// one lane per (b,d) channel; 16 states in registers.
// B/C (32 floats/step, shared block-wide) staged in LDS; with the CDNA5 async
// path, step l+1's B/C are fetched by GLOBAL_LOAD_ASYNC_TO_LDS_B32 (ASYNCcnt)
// while step l computes -> double-buffered, one barrier per step.
// grid = (D_INNER/256, B_SZ), block = 256 (whole block shares b).
__global__ void scan_kernel(const float* __restrict__ dt,
                            const float* __restrict__ xc,
                            const float* __restrict__ xz,      // z = cols [D_INNER, E2)
                            const float* __restrict__ xdbl,    // [T, 96]
                            const float* __restrict__ A_log,
                            const float* __restrict__ Dp,
                            bf16_t* __restrict__ ybf) {
  const int tid = threadIdx.x;
  const int d = blockIdx.x * blockDim.x + tid;
  const int b = blockIdx.y;

  float nA[D_STATE];
  #pragma unroll
  for (int s = 0; s < D_STATE; ++s) nA[s] = -__expf(A_log[d * D_STATE + s]);
  const float Dv = Dp[d];

  float h[D_STATE];
  #pragma unroll
  for (int s = 0; s < D_STATE; ++s) h[s] = 0.0f;

  __shared__ float sBC[2][2 * D_STATE];   // [buffer][B(0..15) | C(16..31)]

#if MAMBA_ASYNC_LDS
  // prologue: async-fetch B/C of step 0 into buffer 0 (wave 0 only)
  if (tid < 2 * D_STATE) {
    g2l_async_b32(&xdbl[(size_t)(b * L_SEQ) * XDBL + DT_RANK + tid], &sBC[0][tid]);
  }
#endif

  for (int l = 0; l < L_SEQ; ++l) {
    const int t = b * L_SEQ + l;
#if MAMBA_ASYNC_LDS
    wait_async0();          // buffer l&1 resident (no-op for non-issuing waves)
    __syncthreads();        // publish LDS; everyone done reading buffer (l+1)&1
    if (l + 1 < L_SEQ && tid < 2 * D_STATE) {
      g2l_async_b32(&xdbl[(size_t)(t + 1) * XDBL + DT_RANK + tid],
                    &sBC[(l + 1) & 1][tid]);
    }
    const float* sB = sBC[l & 1];
#else
    __syncthreads();        // protect sBC before reload
    if (tid < 2 * D_STATE)
      sBC[0][tid] = xdbl[(size_t)t * XDBL + DT_RANK + tid];
    __syncthreads();
    const float* sB = sBC[0];
#endif
    const float* sC = sB + D_STATE;

    const float dtv = dt[(size_t)t * D_INNER + d];
    const float xv  = xc[(size_t)t * D_INNER + d];
    const float dtx = dtv * xv;
    float y = 0.0f;
    #pragma unroll
    for (int s = 0; s < D_STATE; ++s) {
      float dA = __expf(dtv * nA[s]);
      h[s] = h[s] * dA + dtx * sB[s];
      y += h[s] * sC[s];
    }
    const float zv = xz[(size_t)t * E2 + D_INNER + d];
    const float yy = (y + Dv * xv) * (zv * sigmoidf_(zv));
    ybf[(size_t)t * D_INNER + d] = f2bf(yy);
  }
}

// ---------------- residual epilogue (in-place on d_out) ----------------
__global__ void residual_kernel(float* __restrict__ out,
                                const float* __restrict__ seq,
                                const float* __restrict__ scale) {
  int i = blockIdx.x * blockDim.x + threadIdx.x;
  if (i < T_TOK * D_MODEL) out[i] = seq[i] + scale[0] * out[i];
}

// ---------------- host-side orchestration ----------------
extern "C" void kernel_launch(void* const* d_in, const int* in_sizes, int n_in,
                              void* d_out, int out_size, void* d_ws, size_t ws_size,
                              hipStream_t stream) {
  (void)in_sizes; (void)n_in; (void)out_size; (void)ws_size;

  const float* seq       = (const float*)d_in[0];
  const float* ln_g      = (const float*)d_in[1];
  const float* ln_b      = (const float*)d_in[2];
  const float* in_proj_w = (const float*)d_in[3];   // [E2, D_MODEL]
  const float* conv_w    = (const float*)d_in[4];   // [D_INNER, 4]
  const float* conv_b    = (const float*)d_in[5];
  const float* x_proj_w  = (const float*)d_in[6];   // [96, D_INNER]
  const float* dt_proj_w = (const float*)d_in[7];   // [D_INNER, 64]
  const float* dt_proj_b = (const float*)d_in[8];
  const float* A_log     = (const float*)d_in[9];   // [D_INNER, 16]
  const float* Dp        = (const float*)d_in[10];
  const float* out_proj_w= (const float*)d_in[11];  // [D_MODEL, D_INNER]
  const float* res_scale = (const float*)d_in[12];
  float* out = (float*)d_out;

  // --- workspace carving (256B aligned) ---
  char* base = (char*)d_ws;
  size_t off = 0;
  auto carve = [&](size_t bytes) -> char* {
    char* p = base + off;
    off += (bytes + 255) & ~size_t(255);
    return p;
  };
  bf16_t* xn_bf   = (bf16_t*)carve((size_t)T_TOK * D_MODEL * 2);
  bf16_t* w_in_bf = (bf16_t*)carve((size_t)E2 * D_MODEL * 2);
  bf16_t* w_xp_bf = (bf16_t*)carve((size_t)XDBL * D_INNER * 2);
  bf16_t* w_dt_bf = (bf16_t*)carve((size_t)D_INNER * DT_RANK * 2);
  bf16_t* w_out_bf= (bf16_t*)carve((size_t)D_MODEL * D_INNER * 2);
  bf16_t* xc_bf   = (bf16_t*)carve((size_t)T_TOK * D_INNER * 2);
  bf16_t* xdbl_bf = (bf16_t*)carve((size_t)T_TOK * XDBL * 2);
  bf16_t* y_bf    = (bf16_t*)carve((size_t)T_TOK * D_INNER * 2);
  float*  xz      = (float*)carve((size_t)T_TOK * E2 * 4);
  float*  xc      = (float*)carve((size_t)T_TOK * D_INNER * 4);
  float*  xdbl    = (float*)carve((size_t)T_TOK * XDBL * 4);
  float*  dt      = (float*)carve((size_t)T_TOK * D_INNER * 4);

  // --- weight conversions fp32 -> bf16 ---
  auto cvt = [&](const float* s, bf16_t* d, int n) {
    cvt_f32_bf16_kernel<<<(n + 255) / 256, 256, 0, stream>>>(s, d, n);
  };
  cvt(in_proj_w,  w_in_bf,  E2 * D_MODEL);
  cvt(x_proj_w,   w_xp_bf,  XDBL * D_INNER);
  cvt(dt_proj_w,  w_dt_bf,  D_INNER * DT_RANK);
  cvt(out_proj_w, w_out_bf, D_MODEL * D_INNER);

  // --- LayerNorm ---
  ln_kernel<<<T_TOK, 256, 0, stream>>>(seq, ln_g, ln_b, xn_bf);

  // --- in_proj: xz[T,4096] = xn[T,1024] x W_in^T ---
  gemm_bf16_wmma_kernel<<<dim3(E2 / 32, T_TOK / 32), 32, 0, stream>>>(
      xz, nullptr, xn_bf, w_in_bf, nullptr, E2, D_MODEL, D_MODEL, 0);

  // --- depthwise conv + SiLU ---
  conv_silu_kernel<<<(T_TOK * D_INNER + 255) / 256, 256, 0, stream>>>(
      xz, conv_w, conv_b, xc, xc_bf);

  // --- x_proj: xdbl[T,96] = xc[T,2048] x W_xp^T ---
  gemm_bf16_wmma_kernel<<<dim3(XDBL / 32, T_TOK / 32), 32, 0, stream>>>(
      xdbl, xdbl_bf, xc_bf, w_xp_bf, nullptr, XDBL, D_INNER, D_INNER, 0);

  // --- dt_proj + softplus: dt[T,2048] = sp(xdbl[:, :64] x W_dt^T + b) ---
  gemm_bf16_wmma_kernel<<<dim3(D_INNER / 32, T_TOK / 32), 32, 0, stream>>>(
      dt, nullptr, xdbl_bf, w_dt_bf, dt_proj_b, D_INNER, DT_RANK, XDBL, 1);

  // --- selective scan (sequential in L, parallel over B x D_INNER) ---
  scan_kernel<<<dim3(D_INNER / 256, B_SZ), 256, 0, stream>>>(
      dt, xc, xz, xdbl, A_log, Dp, y_bf);

  // --- out_proj: out[T,1024] = y[T,2048] x W_out^T (into d_out) ---
  gemm_bf16_wmma_kernel<<<dim3(D_MODEL / 32, T_TOK / 32), 32, 0, stream>>>(
      out, nullptr, y_bf, w_out_bf, nullptr, D_MODEL, D_INNER, D_INNER, 0);

  // --- residual: out = seq + res_scale * out (in place) ---
  residual_kernel<<<(T_TOK * D_MODEL + 255) / 256, 256, 0, stream>>>(
      out, seq, res_scale);
}
